// IndexedDense_40355512714138
// MI455X (gfx1250) — matrix operand order repaired and checked by reading, not compiled, not run on hardware
//
#include <hip/hip_runtime.h>

// ---- plain clang ext-vector types (trivially copyable, safe to bit-cast) ----
typedef __bf16          bf16x16 __attribute__((ext_vector_type(16)));
typedef float           f32x8   __attribute__((ext_vector_type(8)));
typedef float           f32x4   __attribute__((ext_vector_type(4)));
typedef unsigned int    u32x4   __attribute__((ext_vector_type(4)));
typedef unsigned int    u32x2   __attribute__((ext_vector_type(2)));

// Problem sizes (fixed by the reference)
#define BB 64
#define SS 512
#define DD 1024
#define UU 4096

#define TILE_M 128
#define TILE_N 128
#define TILE_K 32
#define LDA 40    // ushort stride of A tile rows (128 x 32 bf16, padded, 16B-aligned rows)
#define LDB 136   // ushort stride of B tile rows (32 x 128 bf16, padded, 16B-aligned rows)

// Pack two fp32 -> two bf16 with round-to-nearest in 3 VALU ops:
// two v_add (+0x8000) and one v_perm_b32 selecting the high halves.
// dst[15:0] = bf16(a), dst[31:16] = bf16(b)
__device__ __forceinline__ unsigned int pk_bf16(float a, float b) {
    unsigned int ua = __float_as_uint(a) + 0x8000u;
    unsigned int ub = __float_as_uint(b) + 0x8000u;
    // perm({S0=ub, S1=ua}): byte sel 0x02,0x03 -> ua[23:16],ua[31:24]; 0x06,0x07 -> ub[23:16],ub[31:24]
    return __builtin_amdgcn_perm(ub, ua, 0x07060302u);
}

__device__ __forceinline__ u32x2 pk_bf16x4(f32x4 r) {
    u32x2 v;
    v.x = pk_bf16(r.x, r.y);
    v.y = pk_bf16(r.z, r.w);
    return v;
}

struct Pack32 { u32x4 lo; u32x4 hi; };  // 32 bytes == one 16-element bf16 fragment

__global__ __launch_bounds__(256)
void IndexedDense_40355512714138_kernel(const float* __restrict__ x,
                                        const float* __restrict__ ws,
                                        const float* __restrict__ bs,
                                        const int*   __restrict__ index,
                                        float*       __restrict__ out)
{
    __shared__ __align__(16) unsigned short ldsA[TILE_M * LDA];  // [M][K] bf16
    __shared__ __align__(16) unsigned short ldsB[TILE_K * LDB];  // [K][N] bf16

    const int tid  = threadIdx.x;
    const int lane = tid & 31;
    const int wid  = tid >> 5;
    const int wm   = wid & 3;    // 4 wave rows  -> 32-row strip
    const int wn   = wid >> 2;   // 2 wave cols  -> 64-col strip

    const int nBlk = blockIdx.x * TILE_N;
    const int mBlk = blockIdx.y * TILE_M;
    const int b    = blockIdx.z;
    const int e    = index[b];

    const float* A = x  + (size_t)b * SS * DD;   // [S][D], D contiguous
    const float* W = ws + (size_t)e * DD * UU;   // [D][U], U contiguous

    // Global-load work distribution (all float4, fully coalesced)
    const int aCol4 = tid & 7;    // 8 float4 per 32-float A row
    const int aRow0 = tid >> 3;   // 0..31 ; rows aRow0 + 32*i
    const int wCol4 = tid & 31;   // 32 float4 per 128-float W row
    const int wRow0 = tid >> 5;   // 0..7  ; rows wRow0 + 8*i

    f32x4 aReg[4], wReg[4];

    f32x8 acc[2][4];
    #pragma unroll
    for (int i = 0; i < 2; ++i)
        #pragma unroll
        for (int j = 0; j < 4; ++j)
            #pragma unroll
            for (int r = 0; r < 8; ++r) acc[i][j][r] = 0.0f;

    // ---- prologue: fetch K-step 0 into registers ----
    {
        const float* Ap = A + (size_t)mBlk * DD;
        #pragma unroll
        for (int i = 0; i < 4; ++i)
            aReg[i] = *reinterpret_cast<const f32x4*>(Ap + (size_t)(aRow0 + 32 * i) * DD + aCol4 * 4);
        const float* Wp = W + nBlk;
        #pragma unroll
        for (int i = 0; i < 4; ++i)
            wReg[i] = *reinterpret_cast<const f32x4*>(Wp + (size_t)(wRow0 + 8 * i) * UU + wCol4 * 4);
    }

    for (int k0 = 0; k0 < DD; k0 += TILE_K) {
        __syncthreads();  // previous tile fully consumed

        // ---- convert fp32 -> bf16 (v_add + v_perm packing) and stage into LDS (b64 stores) ----
        #pragma unroll
        for (int i = 0; i < 4; ++i) {
            *reinterpret_cast<u32x2*>(&ldsA[(aRow0 + 32 * i) * LDA + aCol4 * 4]) = pk_bf16x4(aReg[i]);
        }
        #pragma unroll
        for (int i = 0; i < 4; ++i) {
            *reinterpret_cast<u32x2*>(&ldsB[(wRow0 + 8 * i) * LDB + wCol4 * 4]) = pk_bf16x4(wReg[i]);
        }

        __syncthreads();  // tile visible to all waves

        // ---- prefetch next K-step while we compute this one ----
        const int kn = k0 + TILE_K;
        if (kn < DD) {
            const float* Ap = A + (size_t)mBlk * DD + kn;
            #pragma unroll
            for (int i = 0; i < 4; ++i)
                aReg[i] = *reinterpret_cast<const f32x4*>(Ap + (size_t)(aRow0 + 32 * i) * DD + aCol4 * 4);
            const float* Wp = W + (size_t)kn * UU + nBlk;
            #pragma unroll
            for (int i = 0; i < 4; ++i)
                wReg[i] = *reinterpret_cast<const f32x4*>(Wp + (size_t)(wRow0 + 8 * i) * UU + wCol4 * 4);
        }

        // ---- build A fragments (ISA layout: lane = M row; K halves split by lane[4]) ----
        // lanes 0-15 : K 0-7 in q[0], K 16-23 in q[1]
        // lanes 16-31: K 8-15 in q[0], K 24-31 in q[1]
        bf16x16 afrag[2];
        {
            const int arow  = lane & 15;
            const int khalf = (lane >> 4) * 8;
            #pragma unroll
            for (int mt = 0; mt < 2; ++mt) {
                const unsigned short* p = &ldsA[(wm * 32 + mt * 16 + arow) * LDA + khalf];
                Pack32 pk;
                pk.lo = *reinterpret_cast<const u32x4*>(p);       // 16B aligned
                pk.hi = *reinterpret_cast<const u32x4*>(p + 16);  // +32B
                afrag[mt] = __builtin_bit_cast(bf16x16, pk);
            }
        }

        // ---- B fragments (lane = N column; K in table order) + WMMA ----
        #pragma unroll
        for (int nt = 0; nt < 4; ++nt) {
            const unsigned short* p = &ldsB[wn * 64 + nt * 16 + (lane & 15)];
            const int kb = (lane >> 4) * 8;
            bf16x16 bfrag;
            #pragma unroll
            for (int s = 0; s < 16; ++s) {
                const int k = (s < 8) ? (kb + s) : (16 + kb + (s - 8));
                bfrag[s] = __builtin_bit_cast(__bf16, p[k * LDB]);
            }
            #pragma unroll
            for (int mt = 0; mt < 2; ++mt) {
                acc[mt][nt] = __builtin_amdgcn_wmma_f32_16x16x32_bf16(
                    /*neg_a=*/false, afrag[mt],
                    /*neg_b=*/false, bfrag,
                    /*c_mod=*/(short)0, acc[mt][nt],
                    /*reuse_a=*/false, /*reuse_b=*/false);
            }
        }
    }

    // ---- epilogue: bias + relu + store (C layout: VGPR r -> M=r (+8 if lane>=16), N=lane&15) ----
    const int col    = lane & 15;
    const int rowOff = (lane >> 4) * 8;
    #pragma unroll
    for (int nt = 0; nt < 4; ++nt) {
        const int n = nBlk + wn * 64 + nt * 16 + col;
        const float bias = bs[(size_t)e * UU + n];
        #pragma unroll
        for (int mt = 0; mt < 2; ++mt) {
            #pragma unroll
            for (int r = 0; r < 8; ++r) {
                const int m = mBlk + wm * 32 + mt * 16 + rowOff + r;
                float v = acc[mt][nt][r] + bias;
                out[((size_t)b * SS + m) * UU + n] = v > 0.0f ? v : 0.0f;
            }
        }
    }
}

extern "C" void kernel_launch(void* const* d_in, const int* in_sizes, int n_in,
                              void* d_out, int out_size, void* d_ws, size_t ws_size,
                              hipStream_t stream) {
    (void)in_sizes; (void)n_in; (void)out_size; (void)d_ws; (void)ws_size;
    const float* x   = (const float*)d_in[0];
    const float* ws  = (const float*)d_in[1];
    const float* bs  = (const float*)d_in[2];
    const int*   idx = (const int*)d_in[3];
    float*       out = (float*)d_out;

    dim3 grid(UU / TILE_N, SS / TILE_M, BB);  // (32, 4, 64)
    dim3 block(256);
    IndexedDense_40355512714138_kernel<<<grid, block, 0, stream>>>(x, ws, bs, idx, out);
}
